// QANet_Att_layer_61778809586096
// MI455X (gfx1250) — matrix operand order repaired and checked by reading, not compiled, not run on hardware
//
#include <hip/hip_runtime.h>

// ---------------------------------------------------------------------------
// QANet-style attention layer for MI455X (gfx1250, wave32, WMMA bf16)
// ---------------------------------------------------------------------------

typedef __attribute__((ext_vector_type(16))) __bf16 v16bf;
typedef __attribute__((ext_vector_type(8)))  __bf16 v8bf;
typedef __attribute__((ext_vector_type(8)))  float  v8f;

namespace {
constexpr int  BC    = 40;      // b_size * num_cand
constexpr int  S     = 128;     // sequence length
constexpr int  H     = 768;     // hidden
constexpr int  NP    = 100;     // num paths
constexpr int  PL    = 20;      // path length
constexpr int  HID   = 100;     // LSTM hidden
constexpr int  G4    = 400;     // 4*HID
constexpr int  NPRE  = 800;     // gates for both directions
constexpr long MPRE  = (long)BC * NP * PL;   // 80000 rows
constexpr int  ENC_LD = 224;    // 2*HID padded to multiple of 32
constexpr int  ROWS  = BC * S;  // 5120
constexpr int  PATHS = BC * NP; // 4000
constexpr int  CATW  = 4 * H;   // 3072
}

// -------------------------- small utility kernels --------------------------

__global__ void k_cvt_bf16(const float* __restrict__ s, __bf16* __restrict__ d, long n) {
    long i = (long)blockIdx.x * blockDim.x + threadIdx.x;
    long stride = (long)gridDim.x * blockDim.x;
    for (; i < n; i += stride) d[i] = (__bf16)s[i];
}

// dst[r][c] = c < scols ? src[r*scols+c] : 0   (bf16, row stride dld)
__global__ void k_pad_cvt(const float* __restrict__ s, __bf16* __restrict__ d,
                          int rows, int scols, int dld) {
    long n = (long)rows * dld;
    long i = (long)blockIdx.x * blockDim.x + threadIdx.x;
    long stride = (long)gridDim.x * blockDim.x;
    for (; i < n; i += stride) {
        int r = (int)(i / dld), c = (int)(i - (long)r * dld);
        d[i] = (c < scols) ? (__bf16)s[(long)r * scols + c] : (__bf16)0.0f;
    }
}

// whhT[k*G4+n] = Whh[n*HID+k]
__global__ void k_transpose_whh(const float* __restrict__ w, float* __restrict__ wt) {
    int i = blockIdx.x * blockDim.x + threadIdx.x;
    if (i >= G4 * HID) return;
    int k = i / G4, n = i - k * G4;
    wt[i] = w[n * HID + k];
}

__global__ void k_bias_cat(const float* bf1, const float* bf2,
                           const float* bb1, const float* bb2, float* out) {
    int i = blockIdx.x * blockDim.x + threadIdx.x;
    if (i >= NPRE) return;
    out[i] = (i < G4) ? (bf1[i] + bf2[i]) : (bb1[i - G4] + bb2[i - G4]);
}

// bert_bf = bf16(bert); w3bert_bf = bf16(bert * W3[h])
__global__ void k_w3bert(const float* __restrict__ bert, const float* __restrict__ w3,
                         __bf16* __restrict__ bert_bf, __bf16* __restrict__ w3bert_bf) {
    long n = (long)ROWS * H;
    long i = (long)blockIdx.x * blockDim.x + threadIdx.x;
    long stride = (long)gridDim.x * blockDim.x;
    for (; i < n; i += stride) {
        int h = (int)(i % H);
        float v = bert[i];
        bert_bf[i] = (__bf16)v;
        w3bert_bf[i] = (__bf16)(v * w3[h]);
    }
}

// ------------------------------ WMMA GEMM ----------------------------------
// C[m][n] (+= bias[n]) = sum_k A[m][k] * B[k][n]
//   A: bf16 row-major [M][K] (lda, K multiple of 32, fully padded/allocated)
//   B: NT=true  -> Bs[n][k] row-major (ldb = padded K)
//        NGUARD=true : rows >= nlim read as 0 (N not multiple of 16)
//        NGUARD=false: all 16 rows of every n-tile are valid storage
//      NT=false -> Bs[k][n] row-major (ldb = N); k >= klim or n >= nlim -> 0
// One 16x16 tile per wave32; 8 waves per block.
template <bool NT, bool NGUARD, bool BIAS, bool RELU, bool WF32, bool WBF>
__global__ void gemm16(const __bf16* __restrict__ A, const __bf16* __restrict__ Bm,
                       const float* __restrict__ bias,
                       float* __restrict__ Cf, __bf16* __restrict__ Cb,
                       int M, int N, int K, int lda, int ldb, int ldc,
                       int nlim, int klim, long sA, long sB, long sC) {
    const int lane = threadIdx.x & 31;
    const int wave = threadIdx.x >> 5;
    const int mtile = blockIdx.y * 8 + wave;
    const int ntile = blockIdx.x;
    if (mtile * 16 >= M) return;

    const int b = blockIdx.z;
    A  += (long)b * sA;
    Bm += (long)b * sB;

    const int half = lane >> 4;      // 0: lanes 0-15, 1: lanes 16-31
    const int l15  = lane & 15;

    v8f acc = {};

    const long arow  = (long)(mtile * 16 + l15) * lda;
    const int  akoff = half * 16;    // A lanes0-15: K 0..15 ; lanes16-31: 16..31
    const int  gn    = ntile * 16 + l15;
    const long bnt   = NT ? ((long)gn * ldb + half * 16) : 0;

    for (int kt = 0; kt < K; kt += 32) {
        v16bf af;
        *((v8bf*)&af)     = *(const v8bf*)(A + arow + kt + akoff);
        *((v8bf*)&af + 1) = *(const v8bf*)(A + arow + kt + akoff + 8);
        if (kt + 32 < K) __builtin_prefetch(A + arow + kt + 64, 0, 1);

        v16bf bf;
        if (NT) {
            if (!NGUARD || gn < nlim) {
                *((v8bf*)&bf)     = *(const v8bf*)(Bm + bnt + kt);
                *((v8bf*)&bf + 1) = *(const v8bf*)(Bm + bnt + kt + 8);
            } else {
                #pragma unroll
                for (int j = 0; j < 16; ++j) bf[j] = (__bf16)0.0f;
            }
        } else {
            const int kb = kt + half * 16;  // B lanes0-15: K 0..15, lanes16-31: 16..31
            #pragma unroll
            for (int j = 0; j < 8; ++j) {
                int k0 = kb + 2 * j, k1 = k0 + 1;
                bf[2 * j]     = (k0 < klim && gn < nlim) ? Bm[(long)k0 * ldb + gn] : (__bf16)0.0f;
                bf[2 * j + 1] = (k1 < klim && gn < nlim) ? Bm[(long)k1 * ldb + gn] : (__bf16)0.0f;
            }
        }
        acc = __builtin_amdgcn_wmma_f32_16x16x32_bf16(false, af, false, bf,
                                                      (short)0, acc, false, false);
    }

    if (gn < nlim) {
        const float bs = BIAS ? bias[gn] : 0.0f;
        const int m0 = mtile * 16 + half * 8;   // C VGPR r: lanes0-15 -> M=r, lanes16-31 -> M=8+r
        const long cb = (long)b * sC;
        #pragma unroll
        for (int r = 0; r < 8; ++r) {
            float v = acc[r] + bs;
            if (RELU) v = v > 0.0f ? v : 0.0f;
            const long off = cb + (long)(m0 + r) * ldc + gn;
            if (WF32) Cf[off] = v;
            if (WBF)  Cb[off] = (__bf16)v;
        }
    }
}

template <bool NT, bool NGUARD, bool BIAS, bool RELU, bool WF32, bool WBF>
static void run_gemm(hipStream_t st, const __bf16* A, const __bf16* B, const float* bias,
                     float* Cf, __bf16* Cb, int M, int N, int K, int lda, int ldb, int ldc,
                     int nlim, int klim, long sA, long sB, long sC, int batch) {
    dim3 g((N + 15) / 16, (M / 16 + 7) / 8, batch);
    gemm16<NT, NGUARD, BIAS, RELU, WF32, WBF><<<g, 256, 0, st>>>(
        A, B, bias, Cf, Cb, M, N, K, lda, ldb, ldc, nlim, klim, sA, sB, sC);
}

// ----------------------------- LSTM recurrence -----------------------------

__device__ __forceinline__ float sigf(float x) { return 1.0f / (1.0f + __expf(-x)); }

// 16 paths per block, one direction per blockIdx.y. pre: [80000][800] f32.
__global__ void k_lstm(const float* __restrict__ pre, const float* __restrict__ whhT,
                       __bf16* __restrict__ enc) {
    __shared__ float h[16 * HID];
    __shared__ float c[16 * HID];
    __shared__ float g[16 * G4];
    const int dir = blockIdx.y;
    const int p0 = blockIdx.x * 16;
    const float* __restrict__ W = whhT + (long)dir * HID * G4;  // [k][n] = Whh[n][k]
    const int tid = threadIdx.x;

    for (int i = tid; i < 16 * HID; i += 256) { h[i] = 0.0f; c[i] = 0.0f; }
    __syncthreads();

    for (int t = 0; t < PL; ++t) {
        const int te = dir ? (PL - 1 - t) : t;
        for (int idx = tid; idx < 16 * G4; idx += 256) {
            const int m = idx / G4, n = idx - m * G4;
            const float* __restrict__ hp = &h[m * HID];
            float s = pre[((long)(p0 + m) * PL + te) * NPRE + dir * G4 + n];
            #pragma unroll 4
            for (int k = 0; k < HID; ++k) s += hp[k] * W[k * G4 + n];
            g[idx] = s;
        }
        __syncthreads();
        for (int idx = tid; idx < 16 * HID; idx += 256) {
            const int m = idx / HID, hh = idx - m * HID;
            const float gi = g[m * G4 + hh];
            const float gf = g[m * G4 + HID + hh];
            const float gg = g[m * G4 + 2 * HID + hh];
            const float go = g[m * G4 + 3 * HID + hh];
            const float cc = sigf(gf) * c[idx] + sigf(gi) * tanhf(gg);
            c[idx] = cc;
            h[idx] = sigf(go) * tanhf(cc);
        }
        __syncthreads();
    }
    for (int idx = tid; idx < 16 * HID; idx += 256) {
        const int m = idx / HID, hh = idx - m * HID;
        enc[(long)(p0 + m) * ENC_LD + dir * HID + hh] = (__bf16)h[idx];
    }
    if (dir == 0) {  // zero the K-padding columns 200..223
        for (int idx = tid; idx < 16 * (ENC_LD - 2 * HID); idx += 256) {
            const int m = idx / (ENC_LD - 2 * HID), cidx = idx % (ENC_LD - 2 * HID);
            enc[(long)(p0 + m) * ENC_LD + 2 * HID + cidx] = (__bf16)0.0f;
        }
    }
}

// ------------------------- attention score pieces --------------------------

__global__ void k_part1(const float* __restrict__ bert, const float* __restrict__ w,
                        const float* __restrict__ b0, float* __restrict__ out) {
    const int row = blockIdx.x * 8 + (threadIdx.x >> 5);
    if (row >= ROWS) return;
    const int lane = threadIdx.x & 31;
    const float* r = bert + (long)row * H;
    float s = 0.0f;
    for (int k = lane; k < H; k += 32) s += r[k] * w[k];
    for (int o = 16; o; o >>= 1) s += __shfl_xor(s, o, 32);
    if (lane == 0) out[row] = s + b0[0];
}

__global__ void k_part2(const __bf16* __restrict__ cs, const float* __restrict__ w,
                        const float* __restrict__ b0, float* __restrict__ out) {
    const int row = blockIdx.x * 8 + (threadIdx.x >> 5);
    if (row >= PATHS) return;
    const int lane = threadIdx.x & 31;
    const __bf16* r = cs + (long)row * H;
    float s = 0.0f;
    for (int k = lane; k < H; k += 32) s += (float)r[k] * w[k];
    for (int o = 16; o; o >>= 1) s += __shfl_xor(s, o, 32);
    if (lane == 0) out[row] = s + b0[0];
}

__global__ void k_total(float* __restrict__ totals, const float* __restrict__ p1,
                        const float* __restrict__ p2, const int* __restrict__ am,
                        const int* __restrict__ cm) {
    const long n = (long)BC * S * NP;
    long i = (long)blockIdx.x * blockDim.x + threadIdx.x;
    if (i >= n) return;
    const int p = (int)(i % NP);
    const long t = i / NP;
    const int s_ = (int)(t % S);
    const int b = (int)(t / S);
    const float mcs = (1.0f - (float)cm[b * NP + p]) * -10000.0f;
    const float mdq = (1.0f - (float)am[b * S + s_]) * -10000.0f;
    totals[i] += p1[b * S + s_] + p2[b * NP + p] + mcs + mdq;
}

// softmax over p (last axis), one wave per (b,s); writes padded [128] bf16 rows
__global__ void k_softmax_p(const float* __restrict__ totals, __bf16* __restrict__ att) {
    const int wid = blockIdx.x * 8 + (threadIdx.x >> 5);
    if (wid >= BC * S) return;
    const int lane = threadIdx.x & 31;
    const float* row = totals + (long)wid * NP;
    float v[4], vmax = -1e30f;
    #pragma unroll
    for (int j = 0; j < 4; ++j) {
        const int p = lane + 32 * j;
        v[j] = (p < NP) ? row[p] : -1e30f;
        vmax = fmaxf(vmax, v[j]);
    }
    for (int o = 16; o; o >>= 1) vmax = fmaxf(vmax, __shfl_xor(vmax, o, 32));
    float sum = 0.0f;
    #pragma unroll
    for (int j = 0; j < 4; ++j) {
        const int p = lane + 32 * j;
        v[j] = (p < NP) ? __expf(v[j] - vmax) : 0.0f;
        sum += v[j];
    }
    for (int o = 16; o; o >>= 1) sum += __shfl_xor(sum, o, 32);
    const float inv = 1.0f / sum;
    __bf16* orow = att + (long)wid * 128;
    #pragma unroll
    for (int j = 0; j < 4; ++j) orow[lane + 32 * j] = (__bf16)(v[j] * inv);
}

// softmax over s (axis 1), one wave per (b,p); pads p>=NP with zeros
__global__ void k_softmax_s(const float* __restrict__ totals, __bf16* __restrict__ att) {
    const int wid = blockIdx.x * 8 + (threadIdx.x >> 5);
    if (wid >= BC * 128) return;
    const int lane = threadIdx.x & 31;
    const int b = wid >> 7, p = wid & 127;
    if (p >= NP) {
        #pragma unroll
        for (int j = 0; j < 4; ++j)
            att[((long)b * 128 + lane + 32 * j) * 128 + p] = (__bf16)0.0f;
        return;
    }
    const float* base = totals + (long)b * S * NP + p;
    float v[4], vmax = -1e30f;
    #pragma unroll
    for (int j = 0; j < 4; ++j) {
        v[j] = base[(long)(lane + 32 * j) * NP];
        vmax = fmaxf(vmax, v[j]);
    }
    for (int o = 16; o; o >>= 1) vmax = fmaxf(vmax, __shfl_xor(vmax, o, 32));
    float sum = 0.0f;
    #pragma unroll
    for (int j = 0; j < 4; ++j) { v[j] = __expf(v[j] - vmax); sum += v[j]; }
    for (int o = 16; o; o >>= 1) sum += __shfl_xor(sum, o, 32);
    const float inv = 1.0f / sum;
    #pragma unroll
    for (int j = 0; j < 4; ++j)
        att[((long)b * 128 + lane + 32 * j) * 128 + p] = (__bf16)(v[j] * inv);
}

// cat = [bert | weight_cs | bert*weight_cs | bert*weighted_context] (bf16)
__global__ void k_cat(const float* __restrict__ bert, const float* __restrict__ wcs,
                      const float* __restrict__ wctx, __bf16* __restrict__ cat) {
    const long n = (long)ROWS * H;
    long i = (long)blockIdx.x * blockDim.x + threadIdx.x;
    long stride = (long)gridDim.x * blockDim.x;
    for (; i < n; i += stride) {
        const long row = i / H;
        const int h = (int)(i - row * H);
        const float b0 = bert[i], wc = wcs[i], wx = wctx[i];
        __bf16* cr = cat + row * CATW;
        cr[h]           = (__bf16)b0;
        cr[H + h]       = (__bf16)wc;
        cr[2 * H + h]   = (__bf16)(b0 * wc);
        cr[3 * H + h]   = (__bf16)(b0 * wx);
    }
}

// x = xbuf + bert; LayerNorm(768) -> out. One wave per row (24 elems/lane).
__global__ void k_ln(const float* __restrict__ xbuf, const float* __restrict__ bert,
                     const float* __restrict__ lw, const float* __restrict__ lb,
                     float* __restrict__ out) {
    const int row = blockIdx.x * 8 + (threadIdx.x >> 5);
    if (row >= ROWS) return;
    const int lane = threadIdx.x & 31;
    const float* xr = xbuf + (long)row * H;
    const float* br = bert + (long)row * H;
    float x[24], s = 0.0f, s2 = 0.0f;
    #pragma unroll
    for (int j = 0; j < 24; ++j) {
        const int h = lane + 32 * j;
        const float v = xr[h] + br[h];
        x[j] = v; s += v; s2 += v * v;
    }
    for (int o = 16; o; o >>= 1) { s += __shfl_xor(s, o, 32); s2 += __shfl_xor(s2, o, 32); }
    const float mean = s * (1.0f / H);
    const float var  = s2 * (1.0f / H) - mean * mean;
    const float inv  = rsqrtf(var + 1e-12f);
    float* orow = out + (long)row * H;
    #pragma unroll
    for (int j = 0; j < 24; ++j) {
        const int h = lane + 32 * j;
        orow[h] = lw[h] * ((x[j] - mean) * inv) + lb[h];
    }
}

// --------------------------------- launcher --------------------------------

extern "C" void kernel_launch(void* const* d_in, const int* in_sizes, int n_in,
                              void* d_out, int out_size, void* d_ws, size_t ws_size,
                              hipStream_t stream) {
    (void)in_sizes; (void)n_in; (void)out_size; (void)ws_size;
    const float* bert  = (const float*)d_in[0];
    const float* csin  = (const float*)d_in[1];
    const int*   amask = (const int*)d_in[2];
    const int*   cmask = (const int*)d_in[3];
    const float* WihF  = (const float*)d_in[8];
    const float* WhhF  = (const float*)d_in[9];
    const float* bihF  = (const float*)d_in[10];
    const float* bhhF  = (const float*)d_in[11];
    const float* WihB  = (const float*)d_in[12];
    const float* WhhB  = (const float*)d_in[13];
    const float* bihB  = (const float*)d_in[14];
    const float* bhhB  = (const float*)d_in[15];
    const float* projW = (const float*)d_in[16];
    const float* projB = (const float*)d_in[17];
    const float* W1w   = (const float*)d_in[18];
    const float* W1b   = (const float*)d_in[19];
    const float* W2w   = (const float*)d_in[20];
    const float* W2b   = (const float*)d_in[21];
    const float* W3    = (const float*)d_in[22];
    const float* attpW = (const float*)d_in[23];
    const float* attpB = (const float*)d_in[24];
    const float* lnw   = (const float*)d_in[25];
    const float* lnb   = (const float*)d_in[26];
    float* out = (float*)d_out;

    // ---- workspace sub-allocation ----
    char* ws = (char*)d_ws;
    size_t off = 0;
    auto take = [&](size_t bytes) -> void* {
        off = (off + 255) & ~(size_t)255;
        void* p = ws + off;
        off += bytes;
        return p;
    };
    // persistent small region
    __bf16* wcat_bf  = (__bf16*)take((size_t)NPRE * H * 2);         // [800][768] = Wih_f|Wih_b
    float*  biascat  = (float*) take((size_t)NPRE * 4);
    __bf16* projW_bf = (__bf16*)take((size_t)H * ENC_LD * 2);       // [768][224] K-padded
    __bf16* attpW_bf = (__bf16*)take((size_t)H * CATW * 2);         // [768][3072]
    __bf16* enc_bf   = (__bf16*)take((size_t)PATHS * ENC_LD * 2);   // [4000][224]
    float*  whhT     = (float*) take((size_t)2 * HID * G4 * 4);     // [dir][100][400]
    float*  part1    = (float*) take((size_t)ROWS * 4);
    float*  part2    = (float*) take((size_t)PATHS * 4);

    // region B: bf16 input copy, later aliased by phase-2 buffers
    off = (off + 255) & ~(size_t)255;
    const size_t offB = off;
    __bf16* csin_bf = (__bf16*)(ws + offB);
    const size_t csinBytes = (size_t)MPRE * H * 2;                  // 122.88 MB

    size_t o2 = offB;  // phase-2 (all consumers of csin_bf finish before these are written)
    auto take2 = [&](size_t bytes) -> void* {
        o2 = (o2 + 255) & ~(size_t)255;
        void* p = ws + o2;
        o2 += bytes;
        return p;
    };
    __bf16* cs_bf     = (__bf16*)take2((size_t)PATHS * H * 2);      // [4000][768]
    __bf16* bert_bf   = (__bf16*)take2((size_t)ROWS * H * 2);
    __bf16* w3bert_bf = (__bf16*)take2((size_t)ROWS * H * 2);
    float*  totals    = (float*) take2((size_t)BC * S * NP * 4);
    __bf16* csatt_bf  = (__bf16*)take2((size_t)BC * S * 128 * 2);   // p padded to 128
    __bf16* ctxatt_bf = (__bf16*)take2((size_t)BC * S * 128 * 2);
    __bf16* probs_bf  = (__bf16*)take2((size_t)BC * S * 128 * 2);
    float*  wcs       = (float*) take2((size_t)ROWS * H * 4);
    float*  wctx      = (float*) take2((size_t)ROWS * H * 4);
    __bf16* cat_bf    = (__bf16*)take2((size_t)ROWS * CATW * 2);
    float*  xbuf      = (float*) take2((size_t)ROWS * H * 4);

    size_t endB = offB + csinBytes;
    if (o2 > endB) endB = o2;
    off = (endB + 255) & ~(size_t)255;
    float* pre = (float*)take((size_t)MPRE * NPRE * 4);             // [80000][800] 256 MB

    // ---- 0) conversions / weight prep ----
    k_cvt_bf16<<<8192, 256, 0, stream>>>(csin, csin_bf, MPRE * (long)H);
    k_cvt_bf16<<<1024, 256, 0, stream>>>(WihF, wcat_bf, (long)G4 * H);
    k_cvt_bf16<<<1024, 256, 0, stream>>>(WihB, wcat_bf + (long)G4 * H, (long)G4 * H);
    k_cvt_bf16<<<4096, 256, 0, stream>>>(attpW, attpW_bf, (long)H * CATW);
    k_bias_cat<<<(NPRE + 255) / 256, 256, 0, stream>>>(bihF, bhhF, bihB, bhhB, biascat);
    k_pad_cvt<<<1024, 256, 0, stream>>>(projW, projW_bf, H, 2 * HID, ENC_LD);
    k_transpose_whh<<<(G4 * HID + 255) / 256, 256, 0, stream>>>(WhhF, whhT);
    k_transpose_whh<<<(G4 * HID + 255) / 256, 256, 0, stream>>>(WhhB, whhT + HID * G4);

    // ---- 1) pre-gates: [80000,768] x [768,800] (B = Wih rows, NT) ----
    run_gemm<true, false, true, false, true, false>(stream, csin_bf, wcat_bf, biascat,
        pre, nullptr, (int)MPRE, NPRE, H, H, H, NPRE, NPRE, H, 0, 0, 0, 1);

    // ---- 2) LSTM recurrence (16 paths/block, dir = blockIdx.y) ----
    {
        dim3 g(PATHS / 16, 2, 1);
        k_lstm<<<g, 256, 0, stream>>>(pre, whhT, enc_bf);
    }

    // ---- 3) cs = relu(enc @ projW^T + b): [4000,224] x [224,768] ----
    run_gemm<true, false, true, true, false, true>(stream, enc_bf, projW_bf, projB,
        nullptr, cs_bf, PATHS, H, ENC_LD, ENC_LD, ENC_LD, H, H, ENC_LD, 0, 0, 0, 1);

    // ---- 4) bert conversions + trilinear scores ----
    k_w3bert<<<4096, 256, 0, stream>>>(bert, W3, bert_bf, w3bert_bf);
    // part3[b]: [128,768] x cs[b]^T (NT, rows = paths, N=100 -> guarded) -> totals
    run_gemm<true, true, false, false, true, false>(stream, w3bert_bf, cs_bf, nullptr,
        totals, nullptr, S, NP, H, H, H, NP, NP, H,
        (long)S * H, (long)NP * H, (long)S * NP, BC);
    k_part1<<<(ROWS + 7) / 8, 256, 0, stream>>>(bert, W1w, W1b, part1);
    k_part2<<<(PATHS + 7) / 8, 256, 0, stream>>>(cs_bf, W2w, W2b, part2);
    k_total<<<((int)((long)BC * S * NP + 255) / 256), 256, 0, stream>>>(
        totals, part1, part2, amask, cmask);

    // ---- 5) softmaxes (bf16, padded to 128 cols) ----
    k_softmax_p<<<(BC * S + 7) / 8, 256, 0, stream>>>(totals, csatt_bf);
    k_softmax_s<<<(BC * 128 + 7) / 8, 256, 0, stream>>>(totals, ctxatt_bf);

    // ---- 6) weight_cs[b] = cs_att[b] @ cs[b] : [128,128pad] x [100,768] (NN) ----
    run_gemm<false, true, false, false, true, false>(stream, csatt_bf, cs_bf, nullptr,
        wcs, nullptr, S, H, 128, 128, H, H, H, NP,
        (long)S * 128, (long)NP * H, (long)S * H, BC);

    // ---- 7) probs[b] = cs_att[b] @ ctx_att[b]^T : [128,128] x [128,128] (NT) ----
    run_gemm<true, false, false, false, false, true>(stream, csatt_bf, ctxatt_bf, nullptr,
        nullptr, probs_bf, S, S, 128, 128, 128, 128, S, 128,
        (long)S * 128, (long)S * 128, (long)S * 128, BC);

    // ---- 8) weighted_context[b] = probs[b] @ bert[b] : [128,128] x [128,768] (NN) ----
    run_gemm<false, true, false, false, true, false>(stream, probs_bf, bert_bf, nullptr,
        wctx, nullptr, S, H, 128, 128, H, H, H, 128,
        (long)S * 128, (long)S * H, (long)S * H, BC);

    // ---- 9) cat + output projection + residual + LayerNorm ----
    k_cat<<<8192, 256, 0, stream>>>(bert, wcs, wctx, cat_bf);
    run_gemm<true, false, true, true, true, false>(stream, cat_bf, attpW_bf, attpB,
        xbuf, nullptr, ROWS, H, CATW, CATW, CATW, H, H, CATW, 0, 0, 0, 1);
    k_ln<<<(ROWS + 7) / 8, 256, 0, stream>>>(xbuf, bert, lnw, lnb, out);
}
